// EmbedEHRLSTMLOS_68109591380521
// MI455X (gfx1250) — matrix operand order repaired and checked by reading, not compiled, not run on hardware
//
#include <hip/hip_runtime.h>
#include <hip/hip_bf16.h>

// Problem constants (from reference)
#define B_  128
#define T_  512
#define L_  8
#define E_  128
#define H_  512
#define G4_ 2048   // 4*H

typedef __attribute__((ext_vector_type(16))) __bf16 bf16x16;
typedef __attribute__((ext_vector_type(8)))  float  f32x8;
typedef __attribute__((ext_vector_type(4)))  unsigned int u32x4;

__device__ __forceinline__ unsigned short f2bf(float f) {
  unsigned u = __float_as_uint(f);
  u += 0x7fffu + ((u >> 16) & 1u);       // round-to-nearest-even
  return (unsigned short)(u >> 16);
}

__device__ __forceinline__ float sigm(float x) {
  return 1.0f / (1.0f + __expf(-x));
}
__device__ __forceinline__ float ftanh(float x) {
  // tanh(x) = 1 - 2/(exp(2x)+1); uses v_exp_f32, branch-free
  return 1.0f - 2.0f / (__expf(2.0f * x) + 1.0f);
}

union FragU { bf16x16 v; u32x4 u[2]; };

// A-fragment (16x32 bf16): lane holds row m; K chunks [kb..kb+7] and [kb+16..kb+23]
__device__ __forceinline__ bf16x16 load_a(const unsigned short* p) {
  FragU r;
  r.u[0] = *(const u32x4*)(p);
  r.u[1] = *(const u32x4*)(p + 16);
  return r.v;
}
// B-fragment (32x16 bf16): lane holds column n; K chunks [kb..kb+7] and [kb+16..kb+23]
__device__ __forceinline__ bf16x16 load_b(const unsigned short* p) {
  FragU r;
  r.u[0] = *(const u32x4*)(p);
  r.u[1] = *(const u32x4*)(p + 16);
  return r.v;
}

// ---------------- grid-wide sense barrier (agent scope) ----------------
__device__ __forceinline__ void grid_barrier(int* cnt, int* gen, int nwg) {
  __syncthreads();
  if (threadIdx.x == 0) {
    __threadfence();
    int g = __hip_atomic_load(gen, __ATOMIC_RELAXED, __HIP_MEMORY_SCOPE_AGENT);
    int v = __hip_atomic_fetch_add(cnt, 1, __ATOMIC_ACQ_REL, __HIP_MEMORY_SCOPE_AGENT);
    if (v == nwg - 1) {
      __hip_atomic_store(cnt, 0, __ATOMIC_RELAXED, __HIP_MEMORY_SCOPE_AGENT);
      __hip_atomic_fetch_add(gen, 1, __ATOMIC_ACQ_REL, __HIP_MEMORY_SCOPE_AGENT);
    } else {
      while (__hip_atomic_load(gen, __ATOMIC_ACQUIRE, __HIP_MEMORY_SCOPE_AGENT) == g) {
        __builtin_amdgcn_s_sleep(2);
      }
    }
  }
  __syncthreads();
}

// ---------------- prep: weights->bf16, bias sum, h0=0, out=b_out, sync=0 ----------------
__global__ void prep_kernel(const float* __restrict__ Wih, const float* __restrict__ Whh,
                            const float* __restrict__ bih, const float* __restrict__ bhh,
                            const float* __restrict__ bout,
                            unsigned short* __restrict__ wih_bf, unsigned short* __restrict__ whh_bf,
                            float* __restrict__ bias, unsigned short* __restrict__ hbuf,
                            float* __restrict__ out, int* __restrict__ sync)
{
  const int stride = gridDim.x * blockDim.x;
  const int i0 = blockIdx.x * blockDim.x + threadIdx.x;
  for (int i = i0; i < G4_ * E_;    i += stride) wih_bf[i] = f2bf(Wih[i]);
  for (int i = i0; i < G4_ * H_;    i += stride) whh_bf[i] = f2bf(Whh[i]);
  for (int i = i0; i < G4_;         i += stride) bias[i]   = bih[i] + bhh[i];
  for (int i = i0; i < 2 * B_ * H_; i += stride) hbuf[i]   = 0;
  const float b0 = bout[0];
  for (int i = i0; i < B_ * T_;     i += stride) out[i]    = b0;
  if (i0 < 2) sync[i0] = 0;
}

// ---------------- embedding: x[t][b][e] = relu(sum_l values * emb[token]) ----------------
__global__ void embed_kernel(const int* __restrict__ tokens, const float* __restrict__ values,
                             const float* __restrict__ emb, unsigned short* __restrict__ x_bf)
{
  const int bt = blockIdx.x;           // bt = b*T + t
  const int b  = bt / T_;
  const int t  = bt - b * T_;
  const int e  = threadIdx.x;          // 0..127
  float acc = 0.0f;
#pragma unroll
  for (int l = 0; l < L_; ++l) {
    const int   tok = tokens[bt * L_ + l];
    const float v   = values[bt * L_ + l];
    acc = fmaf(v, emb[(size_t)tok * E_ + e], acc);
  }
  acc = fmaxf(acc, 0.0f);
  x_bf[((size_t)t * B_ + b) * E_ + e] = f2bf(acc);
}

// ---------------- persistent LSTM, bf16 WMMA, weights pinned in VGPRs ----------------
// grid = 64 WGs x 128 threads (4 wave32 = 1 wave/SIMD at this VGPR usage). WG
// (mgrp,jgrp) owns rows [mgrp*64,+64) x hidden units [jgrp*16,+16). Each wave owns 16
// rows and all 4 gates of its block, so pointwise + c-state + h write + W_out partial
// are wave-local. B fragments (W_ih, W_hh slices, ~640 VGPRs) are loaded ONCE before
// the time loop. Per step: 40 A-fragment loads issued together (one latency exposure),
// then 80 back-to-back v_wmma_f32_16x16x32_bf16.
__global__ void __launch_bounds__(128)
lstm_kernel(const unsigned short* __restrict__ x_bf,
            const unsigned short* __restrict__ wih,
            const unsigned short* __restrict__ whh,
            const float* __restrict__ bias,
            const float* __restrict__ wout,
            unsigned short* __restrict__ hbuf,
            float* __restrict__ out,
            int* sync, int nwg)
{
  const int lane = threadIdx.x & 31;
  const int wv   = threadIdx.x >> 5;      // 0..3
  const int mgrp = blockIdx.x & 1;        // batch-row half
  const int j0   = (blockIdx.x >> 1) << 4;// hidden slice base (0..496)
  const int n    = lane & 15;             // tile column
  const int hi   = lane >> 4;             // half-wave
  const int mrow = mgrp * 64 + wv * 16 + n;       // A-fragment row (= batch index)
  const int mlo  = mgrp * 64 + wv * 16 + hi * 8;  // C/D base row for this lane

  const float bi  = bias[0 * H_ + j0 + n];
  const float bff = bias[1 * H_ + j0 + n];
  const float bg  = bias[2 * H_ + j0 + n];
  const float bo  = bias[3 * H_ + j0 + n];
  const float wo  = wout[j0 + n];

  // ---- Pin all B fragments in registers for the whole sequence ----
  // gates = X*W^T: column (g*H+j0+n) of B is a contiguous K-run of row-major W.
  bf16x16 Bih[4][4];    // [kt][g]  : 128 VGPRs
  bf16x16 Bhh[16][4];   // [kt][g]  : 512 VGPRs
#pragma unroll
  for (int g = 0; g < 4; ++g) {
    const unsigned short* pih = wih + (size_t)(g * H_ + j0 + n) * E_ + hi * 8;
    const unsigned short* phh = whh + (size_t)(g * H_ + j0 + n) * H_ + hi * 8;
#pragma unroll
    for (int kt = 0; kt < 4; ++kt)  Bih[kt][g] = load_b(pih + kt * 32);
#pragma unroll
    for (int kt = 0; kt < 16; ++kt) Bhh[kt][g] = load_b(phh + kt * 32);
  }

  float c[8];
#pragma unroll
  for (int r = 0; r < 8; ++r) c[r] = 0.0f;

  const f32x8 vzero = {0.f, 0.f, 0.f, 0.f, 0.f, 0.f, 0.f, 0.f};

  for (int t = 0; t < T_; ++t) {
    const unsigned short* xt = x_bf + (size_t)t * (B_ * E_) + (size_t)mrow * E_ + hi * 8;
    const unsigned short* hp = hbuf + (size_t)(t & 1) * (B_ * H_) + (size_t)mrow * H_ + hi * 8;
    unsigned short*       hn = hbuf + (size_t)((t + 1) & 1) * (B_ * H_);

    // ---- Issue ALL A-fragment loads for this step before any wmma ----
    bf16x16 ax[4];
#pragma unroll
    for (int kt = 0; kt < 4; ++kt)  ax[kt] = load_a(xt + kt * 32);
    bf16x16 ah[16];
#pragma unroll
    for (int kt = 0; kt < 16; ++kt) ah[kt] = load_a(hp + kt * 32);

    f32x8 acc[4] = {vzero, vzero, vzero, vzero};

    // x_t @ W_ih^T : K = 128
#pragma unroll
    for (int kt = 0; kt < 4; ++kt) {
#pragma unroll
      for (int g = 0; g < 4; ++g) {
        acc[g] = __builtin_amdgcn_wmma_f32_16x16x32_bf16(
            false, ax[kt], false, Bih[kt][g], (short)0, acc[g], false, false);
      }
    }
    // h_{t-1} @ W_hh^T : K = 512
#pragma unroll
    for (int kt = 0; kt < 16; ++kt) {
#pragma unroll
      for (int g = 0; g < 4; ++g) {
        acc[g] = __builtin_amdgcn_wmma_f32_16x16x32_bf16(
            false, ah[kt], false, Bhh[kt][g], (short)0, acc[g], false, false);
      }
    }

    // Pointwise (torch gate order i,f,g,o), h write, and W_out partial dot.
#pragma unroll
    for (int r = 0; r < 8; ++r) {
      float ig = sigm(acc[0][r] + bi);
      float fg = sigm(acc[1][r] + bff);
      float gg = ftanh(acc[2][r] + bg);
      float og = sigm(acc[3][r] + bo);
      c[r] = fg * c[r] + ig * gg;
      float hv = og * ftanh(c[r]);
      hn[(size_t)(mlo + r) * H_ + j0 + n] = f2bf(hv);

      float p = hv * wo;                       // reduce over the 16 hidden units we own
#pragma unroll
      for (int off = 8; off > 0; off >>= 1)
        p += __shfl_xor(p, off, 32);           // xor of bits 0..3 stays in 16-lane group
      if (n == 0)
        atomicAdd(out + (size_t)(mlo + r) * T_ + t, p);
    }

    if (t + 1 < T_) {
      // Next step's x fragments are independent of h: warm them into cache under
      // the barrier (global_prefetch_b8).
      const unsigned short* xn = x_bf + (size_t)(t + 1) * (B_ * E_) + (size_t)mrow * E_;
      __builtin_prefetch(xn, 0, 0);
      __builtin_prefetch(xn + 64, 0, 0);
      grid_barrier(sync, sync + 1, nwg);
    }
  }
}

// ---------------- launcher ----------------
extern "C" void kernel_launch(void* const* d_in, const int* in_sizes, int n_in,
                              void* d_out, int out_size, void* d_ws, size_t ws_size,
                              hipStream_t stream)
{
  const int*   tokens = (const int*)  d_in[0];
  const float* values = (const float*)d_in[1];
  const float* emb    = (const float*)d_in[2];
  const float* Wih    = (const float*)d_in[3];
  const float* Whh    = (const float*)d_in[4];
  const float* bih    = (const float*)d_in[5];
  const float* bhh    = (const float*)d_in[6];
  const float* Wout   = (const float*)d_in[7];
  const float* bout   = (const float*)d_in[8];
  float* out = (float*)d_out;
  (void)in_sizes; (void)n_in; (void)out_size; (void)ws_size;

  // Workspace carve-out (~19.8 MB total)
  char* ws = (char*)d_ws;
  size_t off = 0;
  auto take = [&](size_t bytes) -> void* {
    off = (off + 255) & ~(size_t)255;
    void* p = ws + off;
    off += bytes;
    return p;
  };
  unsigned short* x_bf   = (unsigned short*)take((size_t)T_ * B_ * E_ * 2); // 16.8 MB
  unsigned short* wih_bf = (unsigned short*)take((size_t)G4_ * E_ * 2);     // 0.5 MB
  unsigned short* whh_bf = (unsigned short*)take((size_t)G4_ * H_ * 2);     // 2.0 MB
  float*          bias   = (float*)take((size_t)G4_ * 4);
  unsigned short* hbuf   = (unsigned short*)take((size_t)2 * B_ * H_ * 2);  // 256 KB
  int*            syncp  = (int*)take(256);

  prep_kernel<<<1024, 256, 0, stream>>>(Wih, Whh, bih, bhh, bout,
                                        wih_bf, whh_bf, bias, hbuf, out, syncp);
  embed_kernel<<<B_ * T_, E_, 0, stream>>>(tokens, values, emb, x_bf);

  const int NWG = 64;   // all resident: 64 WGs x 4 wave32 = 1 wave/SIMD on 64 WGPs
  lstm_kernel<<<NWG, 128, 0, stream>>>(x_bf, wih_bf, whh_bf, bias, Wout, hbuf, out,
                                       syncp, NWG);
}
